// MetricLoss_10754598110035
// MI455X (gfx1250) — matrix operand order repaired and checked by reading, not compiled, not run on hardware
//
#include <hip/hip_runtime.h>
#include <hip/hip_bf16.h>
#include <math.h>

// ---------------------------------------------------------------------------
// MetricLoss (discriminative clustering loss) for MI455X / gfx1250.
//
// B=8, D=32, H=W=512, K=16.  Memory-bound: 2 streaming passes over 268MB of
// embeddings + 8MB labels (~24us floor @ 23.3TB/s).  The segment-sum
// (cluster sums) is cast as a one-hot GEMM on V_WMMA_F32_16X16X32_F16:
// one A (one-hot) build feeds two WMMAs covering all 32 dims.
// ---------------------------------------------------------------------------

typedef _Float16 v16h __attribute__((ext_vector_type(16)));
typedef float    v8f  __attribute__((ext_vector_type(8)));
typedef float    f4v  __attribute__((ext_vector_type(4)));

#define KCL   16
#define DDIM  32
#define NPIX  (512 * 512)
#define BIMG  8

// Workspace layout (floats):
//   sums      [B*K*D]   = 4096   @ 0
//   counts    [B*K]     = 128    @ 4096
//   means     [B*K*D]   = 4096   @ 4224
//   hingeSums [B*K]     = 128    @ 8320
//   distLoss  [B]       = 8      @ 8448
//   regLoss   [B]       = 8      @ 8456
#define WS_TOTAL 8464

__global__ void ml_zero_ws(float* ws) {
    int i = blockIdx.x * blockDim.x + threadIdx.x;
    if (i < WS_TOTAL) ws[i] = 0.0f;
}

// ---------------------------------------------------------------------------
// Pass 1: cluster sums via one-hot WMMA GEMM + counts via LDS histogram.
// grid = (128, 1, 8 /*images*/), block = 256 (8 waves).
// Each wave per iteration: chunk of 32 pixels ->
//   A  = onehot[16 clusters x 32 pix] (f16), built once via ds_bpermute
//   B0 = X[32 pix x dims 0..15]  (f16), B1 = X[32 pix x dims 16..31]
//   C0 += A*B0 ; C1 += A*B1      (two v_wmma_f32_16x16x32_f16)
// ---------------------------------------------------------------------------
__global__ void ml_segsum_wmma(const float* __restrict__ emb,
                               const int*   __restrict__ lbls,
                               float* __restrict__ sums,
                               float* __restrict__ counts) {
    const int b  = blockIdx.z;
    const int tid  = threadIdx.x;
    const int lane = tid & 31;
    const int wave = tid >> 5;
    const int g = lane >> 4;                   // lane group (0/1)
    const int n = lane & 15;                   // column (dim offset) and A-row id

    __shared__ float lds_counts[KCL];
    if (tid < KCL) lds_counts[tid] = 0.0f;
    __syncthreads();

    const size_t N = NPIX;
    const float* embImg = emb + (size_t)b * DDIM * N;
    const int*   lblImg = lbls + (size_t)b * N;

    const int wavesPerGrid = gridDim.x * (blockDim.x >> 5);
    const int waveId = blockIdx.x * (blockDim.x >> 5) + wave;
    const size_t step = (size_t)wavesPerGrid * 32;

    v8f c0 = {};  // f32 16x16 accumulator, dims 0..15
    v8f c1 = {};  // f32 16x16 accumulator, dims 16..31

    for (size_t p0 = (size_t)waveId * 32; p0 < N; p0 += step) {
        // --- labels of this 32-pixel chunk (one per lane) ---
        int lbl = lblImg[p0 + lane] & (KCL - 1);
        atomicAdd(&lds_counts[lbl], 1.0f);

        // --- A fragment: one-hot [16 x 32], contiguous k, same order as B ---
        // element i (=2v+h) of this lane corresponds to
        //   k = 16*g + i,  row M = n
        v16h a;
        #pragma unroll
        for (int i = 0; i < 16; ++i) {
            int k = (g << 4) + i;
            int klbl = __builtin_amdgcn_ds_bpermute(k << 2, lbl);
            a[i] = (klbl == n) ? (_Float16)1.0f : (_Float16)0.0f;
        }

        // --- B fragments: X[32 pix x 16 dims]; lane holds column n,
        //     rows k = 16*g .. 16*g+15 -> 64 contiguous bytes per lane ---
        const float* src0 = embImg + (size_t)n * N + p0 + (size_t)(g << 4);
        const float* src1 = src0 + (size_t)16 * N;
        __builtin_prefetch((const void*)(src0 + step), 0, 0);
        __builtin_prefetch((const void*)(src1 + step), 0, 0);

        {
            f4v x0 = ((const f4v*)src0)[0];
            f4v x1 = ((const f4v*)src0)[1];
            f4v x2 = ((const f4v*)src0)[2];
            f4v x3 = ((const f4v*)src0)[3];
            v16h bb;
            #pragma unroll
            for (int j = 0; j < 4; ++j) {
                bb[j]      = (_Float16)x0[j];
                bb[4 + j]  = (_Float16)x1[j];
                bb[8 + j]  = (_Float16)x2[j];
                bb[12 + j] = (_Float16)x3[j];
            }
            c0 = __builtin_amdgcn_wmma_f32_16x16x32_f16(
                    false, a, false, bb, (short)0, c0, false, false);
        }
        {
            f4v x0 = ((const f4v*)src1)[0];
            f4v x1 = ((const f4v*)src1)[1];
            f4v x2 = ((const f4v*)src1)[2];
            f4v x3 = ((const f4v*)src1)[3];
            v16h bb;
            #pragma unroll
            for (int j = 0; j < 4; ++j) {
                bb[j]      = (_Float16)x0[j];
                bb[4 + j]  = (_Float16)x1[j];
                bb[8 + j]  = (_Float16)x2[j];
                bb[12 + j] = (_Float16)x3[j];
            }
            c1 = __builtin_amdgcn_wmma_f32_16x16x32_f16(
                    false, a, false, bb, (short)0, c1, false, false);
        }
    }

    // --- reduce accumulators to global sums; C layout: VGPR r -> M = r + 8*g,
    //     column N = lane & 15 ---
    #pragma unroll
    for (int r = 0; r < 8; ++r) {
        int m = r + (g << 3);
        atomicAdd(&sums[((size_t)b * KCL + m) * DDIM + n],      c0[r]);
        atomicAdd(&sums[((size_t)b * KCL + m) * DDIM + 16 + n], c1[r]);
    }
    __syncthreads();
    if (tid < KCL) atomicAdd(&counts[b * KCL + tid], lds_counts[tid]);
}

// ---------------------------------------------------------------------------
// Pass 2: means = sums / max(counts,1); distance (push) loss; reg loss.
// grid = (8) one block per image, block = 256.
// ---------------------------------------------------------------------------
__global__ void ml_means_dist(const float* __restrict__ sums,
                              const float* __restrict__ counts,
                              float* __restrict__ means,
                              float* __restrict__ distLoss,
                              float* __restrict__ regLoss) {
    const int b = blockIdx.x;
    const int tid = threadIdx.x;
    __shared__ float sMean[KCL * DDIM];
    __shared__ float red[256];

    for (int idx = tid; idx < KCL * DDIM; idx += blockDim.x) {
        int k = idx >> 5;
        float cnt = fmaxf(counts[b * KCL + k], 1.0f);
        float m = sums[(size_t)b * KCL * DDIM + idx] / cnt;
        sMean[idx] = m;
        means[(size_t)b * KCL * DDIM + idx] = m;
    }
    __syncthreads();

    // pairwise hinge over unordered pairs i<j (thread t -> (i,j)=(t>>4,t&15))
    const int i = tid >> 4, j = tid & 15;
    float contrib = 0.0f;
    if (i < j) {
        float sq = 0.0f;
        #pragma unroll
        for (int d = 0; d < DDIM; ++d) {
            float df = sMean[i * DDIM + d] - sMean[j * DDIM + d];
            sq += df * df;
        }
        float h = fmaxf(2.0f * 1.5f - sqrtf(sq), 0.0f);
        contrib = h * h;
    }
    red[tid] = contrib;
    __syncthreads();
    for (int s = 128; s > 0; s >>= 1) {
        if (tid < s) red[tid] += red[tid + s];
        __syncthreads();
    }
    if (tid == 0) distLoss[b] = red[0] / (float)(KCL * (KCL - 1));
    __syncthreads();

    // regularization: mean L2 norm of cluster means (threads with i==j)
    float regc = 0.0f;
    if (i == j) {
        float sq = 0.0f;
        #pragma unroll
        for (int d = 0; d < DDIM; ++d) {
            float m = sMean[i * DDIM + d];
            sq += m * m;
        }
        regc = sqrtf(sq);
    }
    red[tid] = regc;
    __syncthreads();
    for (int s = 128; s > 0; s >>= 1) {
        if (tid < s) red[tid] += red[tid + s];
        __syncthreads();
    }
    if (tid == 0) regLoss[b] = red[0] / (float)KCL;
}

// ---------------------------------------------------------------------------
// Pass 3: variance (pull) loss. Streams embeddings once more: 4 pixels per
// lane via b128 non-temporal loads (last use -> don't pollute L2); means
// LDS-resident; per-cluster hinge sums via LDS f32 atomics.
// grid = (128, 1, 8), block = 256.
// ---------------------------------------------------------------------------
__global__ void ml_variance(const float* __restrict__ emb,
                            const int*   __restrict__ lbls,
                            const float* __restrict__ means,
                            float* __restrict__ hingeSums) {
    const int b = blockIdx.z;
    const int tid = threadIdx.x;
    __shared__ float sMean[KCL * DDIM];
    __shared__ float sH[KCL];

    for (int idx = tid; idx < KCL * DDIM; idx += blockDim.x)
        sMean[idx] = means[(size_t)b * KCL * DDIM + idx];
    if (tid < KCL) sH[tid] = 0.0f;
    __syncthreads();

    const size_t N  = NPIX;
    const size_t Q  = NPIX / 4;                 // pixel quads
    const float* e  = emb + (size_t)b * DDIM * N;
    const int*   l  = lbls + (size_t)b * N;
    const size_t stride = (size_t)gridDim.x * blockDim.x;

    for (size_t q = (size_t)blockIdx.x * blockDim.x + tid; q < Q; q += stride) {
        int4 l4 = ((const int4*)l)[q];
        const int la = l4.x & (KCL - 1), lb = l4.y & (KCL - 1);
        const int lc = l4.z & (KCL - 1), ld = l4.w & (KCL - 1);
        float d2a = 0.0f, d2b = 0.0f, d2c = 0.0f, d2d = 0.0f;
        #pragma unroll
        for (int d = 0; d < DDIM; ++d) {
            f4v x = __builtin_nontemporal_load((const f4v*)(e + (size_t)d * N) + q);
            float va = x[0] - sMean[la * DDIM + d];
            float vb = x[1] - sMean[lb * DDIM + d];
            float vc = x[2] - sMean[lc * DDIM + d];
            float vd = x[3] - sMean[ld * DDIM + d];
            d2a += va * va; d2b += vb * vb; d2c += vc * vc; d2d += vd * vd;
        }
        float ta = fmaxf(sqrtf(d2a) - 0.5f, 0.0f);   // VAR_THETA = 0.5
        float tb = fmaxf(sqrtf(d2b) - 0.5f, 0.0f);
        float tc = fmaxf(sqrtf(d2c) - 0.5f, 0.0f);
        float td = fmaxf(sqrtf(d2d) - 0.5f, 0.0f);
        atomicAdd(&sH[la], ta * ta);
        atomicAdd(&sH[lb], tb * tb);
        atomicAdd(&sH[lc], tc * tc);
        atomicAdd(&sH[ld], td * td);
    }
    __syncthreads();
    if (tid < KCL) atomicAdd(&hingeSums[b * KCL + tid], sH[tid]);
}

// ---------------------------------------------------------------------------
// Pass 4: combine into scalar loss.
// ---------------------------------------------------------------------------
__global__ void ml_finalize(const float* __restrict__ counts,
                            const float* __restrict__ hingeSums,
                            const float* __restrict__ distLoss,
                            const float* __restrict__ regLoss,
                            float* __restrict__ out) {
    __shared__ float red[BIMG];
    const int tid = threadIdx.x;
    if (tid < BIMG) {
        float s = 0.0f;
        for (int k = 0; k < KCL; ++k)
            s += hingeSums[tid * KCL + k] / fmaxf(counts[tid * KCL + k], 1.0f);
        float var = s / (float)KCL;
        red[tid] = 1.0f * var + 1.0f * distLoss[tid] + 0.001f * regLoss[tid];
    }
    __syncthreads();
    if (tid == 0) {
        float t = 0.0f;
        for (int i = 0; i < BIMG; ++i) t += red[i];
        out[0] = t / (float)BIMG;
    }
}

// ---------------------------------------------------------------------------
extern "C" void kernel_launch(void* const* d_in, const int* in_sizes, int n_in,
                              void* d_out, int out_size, void* d_ws, size_t ws_size,
                              hipStream_t stream) {
    const float* emb  = (const float*)d_in[0];   // [8, 32, 512, 512] f32
    const int*   lbls = (const int*)d_in[1];     // [8, 512, 512] i32
    float* out = (float*)d_out;

    float* ws        = (float*)d_ws;
    float* sums      = ws;            // 4096
    float* counts    = ws + 4096;     // 128
    float* means     = ws + 4224;     // 4096
    float* hingeSums = ws + 8320;     // 128
    float* distLoss  = ws + 8448;     // 8
    float* regLoss   = ws + 8456;     // 8

    ml_zero_ws<<<(WS_TOTAL + 255) / 256, 256, 0, stream>>>(ws);

    ml_segsum_wmma<<<dim3(128, 1, BIMG), 256, 0, stream>>>(emb, lbls, sums, counts);

    ml_means_dist<<<dim3(BIMG), 256, 0, stream>>>(sums, counts, means, distLoss, regLoss);

    ml_variance<<<dim3(128, 1, BIMG), 256, 0, stream>>>(emb, lbls, means, hingeSums);

    ml_finalize<<<dim3(1), 64, 0, stream>>>(counts, hingeSums, distLoss, regLoss, out);
}
